// MoE10DimBlock_24550033064020
// MI455X (gfx1250) — matrix operand order repaired and checked by reading, not compiled, not run on hardware
//
#include <hip/hip_runtime.h>

// ---------------------------------------------------------------------------
// Problem constants (from reference)
// ---------------------------------------------------------------------------
constexpr int   kB    = 4;
constexpr int   kT    = 1024;
constexpr int   kD    = 1024;
constexpr int   kH    = 16;
constexpr int   kHKV  = 4;
constexpr int   kHD   = 64;
constexpr int   kDPF  = 4096;
constexpr int   kDEXP = 2048;
constexpr int   kNEXP = 10;
constexpr int   kTOPK = 3;
constexpr int   kM    = kB * kT;        // 4096 rows
constexpr int   kKVD  = kHKV * kHD;     // 256
constexpr float kGS   = 0.001f;
constexpr float kEPS  = 1e-6f;

// ---------------------------------------------------------------------------
// Vector types (wave32, CDNA5)
// ---------------------------------------------------------------------------
typedef __attribute__((ext_vector_type(16))) __bf16 v16bf;
typedef __attribute__((ext_vector_type(8)))  __bf16 v8bf;
typedef __attribute__((ext_vector_type(8)))  float  v8f;

__device__ __forceinline__ __bf16 f2bf(float f) {
  unsigned u = __builtin_bit_cast(unsigned, f);
  unsigned r = u + 0x7fffu + ((u >> 16) & 1u);   // round-to-nearest-even
  unsigned short s = (unsigned short)(r >> 16);
  return __builtin_bit_cast(__bf16, s);
}

__device__ __forceinline__ v16bf cat8(v8bf lo, v8bf hi) {
  return __builtin_shufflevector(lo, hi, 0,1,2,3,4,5,6,7,8,9,10,11,12,13,14,15);
}

// A fragment (16x32 bf16): lanes 0-15 hold K 0-7 / 16-23, lanes 16-31 hold
// K 8-15 / 24-31 (CDNA5 ISA 7.12.2). base = row pointer into LDS tile.
__device__ __forceinline__ v16bf load_frag_a(const __bf16* base, int hv) {
  v8bf lo = *(const v8bf*)(base + hv * 8);
  v8bf hi = *(const v8bf*)(base + 16 + hv * 8);
  return cat8(lo, hi);
}
// B fragment (32x16 bf16, N-major LDS): lanes 0-15 hold K 0-7 / 16-23,
// lanes 16-31 hold K 8-15 / 24-31.
__device__ __forceinline__ v16bf load_frag_b(const __bf16* base, int hv) {
  v8bf lo = *(const v8bf*)(base + hv * 8);
  v8bf hi = *(const v8bf*)(base + 16 + hv * 8);
  return cat8(lo, hi);
}

// Async global->LDS copy of 16 bytes per lane (ASYNCcnt-tracked, CDNA5).
__device__ __forceinline__ unsigned ldsoff(const void* p) {
  return (unsigned)(size_t)p;   // generic LDS pointer: addr[31:0] = LDS offset
}
__device__ __forceinline__ void async_b128(unsigned lds, const void* g) {
  asm volatile("global_load_async_to_lds_b128 %0, %1, off"
               :: "v"(lds), "v"((unsigned long long)(size_t)g) : "memory");
}
__device__ __forceinline__ void wait_async0() {
  asm volatile("s_wait_asynccnt 0x0" ::: "memory");
}

// ---------------------------------------------------------------------------
// One-time per-call weight pass: f32 [K,N] -> bf16 transposed [N,K].
// Streams every weight once (~480 MB @ 23.3 TB/s ~ 21 us) so all GEMM hot
// loops become conversion-free raw bf16 copies with half the HBM traffic.
// ---------------------------------------------------------------------------
__global__ __launch_bounds__(256)
void transpose_cvt_kernel(const float* __restrict__ in, __bf16* __restrict__ out,
                          int K, int N) {
  __shared__ float tile[32][33];
  const size_t mstride = (size_t)K * N;
  const float* src = in + (size_t)blockIdx.z * mstride;
  __bf16* dst = out + (size_t)blockIdx.z * mstride;
  int k0 = blockIdx.y * 32, n0 = blockIdx.x * 32;
  int tx = threadIdx.x & 31, ty = threadIdx.x >> 5;   // 32x8
#pragma unroll
  for (int j = 0; j < 32; j += 8)
    tile[ty + j][tx] = src[(size_t)(k0 + ty + j) * N + n0 + tx];
  __syncthreads();
#pragma unroll
  for (int j = 0; j < 32; j += 8)
    dst[(size_t)(n0 + ty + j) * K + k0 + tx] = f2bf(tile[tx][ty + j]);
}

// ---------------------------------------------------------------------------
// Tiled bf16 WMMA GEMM:  C[M,N] (+epilogue) = A[M,K] @ Wt[N,K]^T
// A, Wt are bf16; Wt is pre-transposed (N-major). 256 threads = 8 waves
// (4xM, 2xN). BM=BN=128, BK=64 (two WMMA k-steps per stage). Double-buffered
// LDS filled with global_load_async_to_lds_b128, synced via s_wait_asynccnt.
// MODE 0: C = acc                        (OUTBF may be true -> bf16 store)
// MODE 1: C = R + acc                    (residual fuse)
// MODE 2: C = R + acc + kGS * CS         (pf_down epilogue)
// MODE 3: C = R + scale_b * acc          (MoE down accumulate, R may alias C)
// INDEXED: Wt base += eidx[b*stride]*K*N (per-batch expert weights)
// ---------------------------------------------------------------------------
constexpr int GBM = 128, GBN = 128, GBK = 64, GLS = 72;  // GLS: padded stride

template <int MODE, bool INDEXED, bool OUTBF>
__global__ __launch_bounds__(256)
void gemm_bf16_wmma(const __bf16* __restrict__ A, const __bf16* __restrict__ Wt,
                    void* __restrict__ Cout, int M, int N, int K,
                    const float* __restrict__ R, const float* __restrict__ CS,
                    const int* __restrict__ eidx, const float* __restrict__ scales,
                    int sstride, int Tb) {
  __shared__ __bf16 As[2][GBM][GLS];
  __shared__ __bf16 Bs[2][GBN][GLS];   // N-major: Bs[buf][n][k]

  const int tid  = threadIdx.x;
  const int wid  = tid >> 5;
  const int lane = tid & 31;
  const int wm   = wid & 3;            // wave tile row (x32 rows)
  const int wn   = wid >> 2;           // wave tile col (x64 cols)
  const int hv   = lane >> 4;
  const int ln   = lane & 15;
  const int rowBase = blockIdx.y * GBM;
  const int colBase = blockIdx.x * GBN;
  const int b = rowBase / Tb;          // batch index (GBM divides Tb)

  const __bf16* Wb = Wt;
  if (INDEXED) Wb = Wt + (size_t)eidx[b * sstride] * (size_t)K * (size_t)N;
  float scale = 1.0f;
  if (MODE == 3) scale = scales[b * sstride];

  // staging indices: 1024 16B-chunks per tile, 4 per thread
  const int sr = tid >> 3;             // will be offset by pass
  const int sc = (tid & 7) * 8;

  v8f acc[2][4];
#pragma unroll
  for (int i = 0; i < 2; ++i)
#pragma unroll
    for (int j = 0; j < 4; ++j)
#pragma unroll
      for (int e = 0; e < 8; ++e) acc[i][j][e] = 0.f;

  auto stage = [&](int buf, int k0) {
#pragma unroll
    for (int p = 0; p < 4; ++p) {
      int r = p * 32 + sr;
      async_b128(ldsoff(&As[buf][r][sc]), A  + (size_t)(rowBase + r) * K + k0 + sc);
      async_b128(ldsoff(&Bs[buf][r][sc]), Wb + (size_t)(colBase + r) * K + k0 + sc);
    }
  };

  stage(0, 0);
  const int nk = K / GBK;
  for (int i = 0; i < nk; ++i) {
    const int cur = i & 1;
    wait_async0();          // this wave's tile-fill for buf[cur] has landed
    __syncthreads();        // all waves filled buf[cur]; prior reads of other buf done
    if (i + 1 < nk) stage(cur ^ 1, (i + 1) * GBK);
#pragma unroll
    for (int kk = 0; kk < 2; ++kk) {
      v16bf a0 = load_frag_a(&As[cur][wm * 32 + ln][kk * 32], hv);
      v16bf a1 = load_frag_a(&As[cur][wm * 32 + 16 + ln][kk * 32], hv);
#pragma unroll
      for (int nt = 0; nt < 4; ++nt) {
        v16bf bb = load_frag_b(&Bs[cur][wn * 64 + nt * 16 + ln][kk * 32], hv);
        acc[0][nt] = __builtin_amdgcn_wmma_f32_16x16x32_bf16(
            false, a0, false, bb, (short)0, acc[0][nt], false, false);
        acc[1][nt] = __builtin_amdgcn_wmma_f32_16x16x32_bf16(
            false, a1, false, bb, (short)0, acc[1][nt], false, false);
      }
    }
  }

  // epilogue: C/D layout -> row = r + 8*hv (+16 per m-subtile), col = ln
#pragma unroll
  for (int mt = 0; mt < 2; ++mt)
#pragma unroll
    for (int nt = 0; nt < 4; ++nt)
#pragma unroll
      for (int r = 0; r < 8; ++r) {
        int row = rowBase + wm * 32 + mt * 16 + r + 8 * hv;
        int col = colBase + wn * 64 + nt * 16 + ln;
        size_t idx = (size_t)row * N + col;
        float v = acc[mt][nt][r];
        if (OUTBF) {
          ((__bf16*)Cout)[idx] = f2bf(v);
        } else {
          float* Cf = (float*)Cout;
          if (MODE == 0)      Cf[idx] = v;
          else if (MODE == 1) Cf[idx] = R[idx] + v;
          else if (MODE == 2) Cf[idx] = R[idx] + v + kGS * CS[idx];
          else                Cf[idx] = R[idx] + scale * v;
        }
      }
}

// ---------------------------------------------------------------------------
// Flash attention (causal, GQA), all-bf16 I/O. 128 threads = 4 waves; each
// wave owns 32 q rows. Q block 128 x 64, K/V blocks of 64 keys. Q/K staged
// with async-to-LDS; V transposed manually (no conversions, inputs are bf16).
// 1/sqrt(HD) applied to S after the QK^T WMMA. Output written in place over q.
// ---------------------------------------------------------------------------
constexpr int FQ = 128, FKV = 64, FST = 72;

__global__ __launch_bounds__(128)
void flash_attn_kernel(const __bf16* __restrict__ q, const __bf16* __restrict__ k,
                       const __bf16* __restrict__ v, __bf16* __restrict__ o) {
  __shared__ __bf16 Qs[FQ][FST];
  __shared__ __bf16 Ks[FKV][FST];
  __shared__ __bf16 Vts[kHD][FST];      // V transposed: Vts[dim][key]
  __shared__ __bf16 Ps[4][32][FST];     // per-wave P scratch

  const int tid = threadIdx.x;
  const int wid = tid >> 5;
  const int lane = tid & 31;
  const int hv = lane >> 4;
  const int ln = lane & 15;
  const int qblk = blockIdx.x;
  const int h = blockIdx.y;
  const int b = blockIdx.z;
  const int hkv = h / (kH / kHKV);
  const int t0 = qblk * FQ;

  // async-stage Q: 1024 16B chunks, 8 per thread
#pragma unroll
  for (int p = 0; p < 8; ++p) {
    int e = p * 128 + tid;
    int r = e >> 3, c = (e & 7) * 8;
    async_b128(ldsoff(&Qs[r][c]),
               q + (((size_t)b * kT + t0 + r) * kH + h) * kHD + c);
  }

  float mrow[2][8], lrow[2][8];
  v8f oacc[2][4];
#pragma unroll
  for (int mt = 0; mt < 2; ++mt)
#pragma unroll
    for (int r = 0; r < 8; ++r) { mrow[mt][r] = -1e30f; lrow[mt][r] = 0.f; }
#pragma unroll
  for (int mt = 0; mt < 2; ++mt)
#pragma unroll
    for (int dt = 0; dt < 4; ++dt)
#pragma unroll
      for (int e = 0; e < 8; ++e) oacc[mt][dt][e] = 0.f;

  const int nkb = (qblk + 1) * (FQ / FKV);
  for (int kb = 0; kb < nkb; ++kb) {
    const int kt0 = kb * FKV;
    __syncthreads();     // previous tile fully consumed before overwrite
    // async-stage K (row-major keys x dim == N-major B layout for Q@K^T)
#pragma unroll
    for (int p = 0; p < 4; ++p) {
      int e = p * 128 + tid;
      int r = e >> 3, c = (e & 7) * 8;
      async_b128(ldsoff(&Ks[r][c]),
                 k + (((size_t)b * kT + kt0 + r) * kHKV + hkv) * kHD + c);
    }
    // stage V transposed (dim-major B layout for P@V); raw bf16, no cvt
#pragma unroll
    for (int p = 0; p < 4; ++p) {
      int e = p * 128 + tid;
      int r = e >> 3, c = (e & 7) * 8;
      v8bf val = *(const v8bf*)(v + (((size_t)b * kT + kt0 + r) * kHKV + hkv) * kHD + c);
#pragma unroll
      for (int j = 0; j < 8; ++j) Vts[c + j][r] = val[j];
    }
    wait_async0();       // this wave's K (and first-iter Q) chunks landed
    __syncthreads();     // everyone staged (compiler waits DS before barrier)

    // S = Q @ K^T  (contract HD=64 in two K=32 steps)
    v8f sacc[2][4];
#pragma unroll
    for (int mt = 0; mt < 2; ++mt)
#pragma unroll
      for (int nt = 0; nt < 4; ++nt)
#pragma unroll
        for (int e = 0; e < 8; ++e) sacc[mt][nt][e] = 0.f;
#pragma unroll
    for (int kk = 0; kk < 2; ++kk) {
      v16bf a0 = load_frag_a(&Qs[wid * 32 + ln][kk * 32], hv);
      v16bf a1 = load_frag_a(&Qs[wid * 32 + 16 + ln][kk * 32], hv);
#pragma unroll
      for (int nt = 0; nt < 4; ++nt) {
        v16bf bb = load_frag_b(&Ks[nt * 16 + ln][kk * 32], hv);
        sacc[0][nt] = __builtin_amdgcn_wmma_f32_16x16x32_bf16(
            false, a0, false, bb, (short)0, sacc[0][nt], false, false);
        sacc[1][nt] = __builtin_amdgcn_wmma_f32_16x16x32_bf16(
            false, a1, false, bb, (short)0, sacc[1][nt], false, false);
      }
    }

    // scale + causal mask + online softmax update
#pragma unroll
    for (int mt = 0; mt < 2; ++mt)
#pragma unroll
      for (int r = 0; r < 8; ++r) {
        int qt = t0 + wid * 32 + mt * 16 + r + 8 * hv;
        float mx = -1e30f;
#pragma unroll
        for (int nt = 0; nt < 4; ++nt) {
          int kt = kt0 + nt * 16 + ln;
          float s = sacc[mt][nt][r] * 0.125f;      // 1/sqrt(64)
          s = (kt <= qt) ? s : -1e30f;
          sacc[mt][nt][r] = s;
          mx = fmaxf(mx, s);
        }
        mx = fmaxf(mx, __shfl_xor(mx, 1, 32));
        mx = fmaxf(mx, __shfl_xor(mx, 2, 32));
        mx = fmaxf(mx, __shfl_xor(mx, 4, 32));
        mx = fmaxf(mx, __shfl_xor(mx, 8, 32));
        float mnew  = fmaxf(mrow[mt][r], mx);
        float alpha = __expf(mrow[mt][r] - mnew);
        mrow[mt][r] = mnew;
        float rs = 0.f;
#pragma unroll
        for (int nt = 0; nt < 4; ++nt) {
          float p_ = __expf(sacc[mt][nt][r] - mnew);
          sacc[mt][nt][r] = p_;
          rs += p_;
        }
        rs += __shfl_xor(rs, 1, 32);
        rs += __shfl_xor(rs, 2, 32);
        rs += __shfl_xor(rs, 4, 32);
        rs += __shfl_xor(rs, 8, 32);
        lrow[mt][r] = lrow[mt][r] * alpha + rs;
#pragma unroll
        for (int dt = 0; dt < 4; ++dt) oacc[mt][dt][r] *= alpha;
        // spill P to per-wave LDS (C layout -> A layout)
#pragma unroll
        for (int nt = 0; nt < 4; ++nt)
          Ps[wid][mt * 16 + r + 8 * hv][nt * 16 + ln] = f2bf(sacc[mt][nt][r]);
      }
    asm volatile("s_wait_dscnt 0x0" ::: "memory");  // wave-local LDS RAW fence

    // O += P @ V  (contract 64 keys in two K=32 steps)
#pragma unroll
    for (int kk = 0; kk < 2; ++kk) {
      v16bf a0 = load_frag_a(&Ps[wid][ln][kk * 32], hv);
      v16bf a1 = load_frag_a(&Ps[wid][16 + ln][kk * 32], hv);
#pragma unroll
      for (int dt = 0; dt < 4; ++dt) {
        v16bf bb = load_frag_b(&Vts[dt * 16 + ln][kk * 32], hv);
        oacc[0][dt] = __builtin_amdgcn_wmma_f32_16x16x32_bf16(
            false, a0, false, bb, (short)0, oacc[0][dt], false, false);
        oacc[1][dt] = __builtin_amdgcn_wmma_f32_16x16x32_bf16(
            false, a1, false, bb, (short)0, oacc[1][dt], false, false);
      }
    }
  }

  // finalize: o = oacc / l (bf16), over the q region owned by this block
#pragma unroll
  for (int mt = 0; mt < 2; ++mt)
#pragma unroll
    for (int dt = 0; dt < 4; ++dt)
#pragma unroll
      for (int r = 0; r < 8; ++r) {
        int qt = t0 + wid * 32 + mt * 16 + r + 8 * hv;
        int d  = dt * 16 + ln;
        o[(((size_t)b * kT + qt) * kH + h) * kHD + d] =
            f2bf(oacc[mt][dt][r] / lrow[mt][r]);
      }
}

// ---------------------------------------------------------------------------
// RMSNorm: one block per row; f32 in, bf16 out (feeds GEMM A operand)
// ---------------------------------------------------------------------------
__global__ __launch_bounds__(256)
void rmsnorm_kernel(const float* __restrict__ x, const float* __restrict__ w,
                    __bf16* __restrict__ out, int Dn) {
  __shared__ float red[256];
  size_t row = blockIdx.x;
  const float* xr = x + row * Dn;
  float ss = 0.f;
  for (int c = threadIdx.x; c < Dn; c += 256) { float t = xr[c]; ss += t * t; }
  red[threadIdx.x] = ss;
  __syncthreads();
  for (int s = 128; s > 0; s >>= 1) {
    if (threadIdx.x < s) red[threadIdx.x] += red[threadIdx.x + s];
    __syncthreads();
  }
  float inv = rsqrtf(red[0] / (float)Dn + kEPS);
  for (int c = threadIdx.x; c < Dn; c += 256)
    out[row * Dn + c] = f2bf(xr[c] * inv * w[c]);
}

// ---------------------------------------------------------------------------
// hid = silu(g) * u, bf16 in/out (in place over g); optional tension (f32)
// ---------------------------------------------------------------------------
__global__ __launch_bounds__(256)
void silu_mul_kernel(const __bf16* __restrict__ g, const __bf16* __restrict__ u,
                     __bf16* __restrict__ out, float* __restrict__ tension, int Nn) {
  __shared__ float red[256];
  size_t row = blockIdx.x;
  const __bf16* gr = g + row * Nn;
  const __bf16* ur = u + row * Nn;
  float ss = 0.f;
  for (int c = threadIdx.x; c < Nn; c += 256) {
    float gv = (float)gr[c];
    float hid = (gv / (1.f + __expf(-gv))) * (float)ur[c];
    out[row * Nn + c] = f2bf(hid);
    ss += hid * hid;
  }
  if (tension != nullptr) {
    red[threadIdx.x] = ss;
    __syncthreads();
    for (int s = 128; s > 0; s >>= 1) {
      if (threadIdx.x < s) red[threadIdx.x] += red[threadIdx.x + s];
      __syncthreads();
    }
    if (threadIdx.x == 0) tension[row] = red[0] / (float)Nn;
  }
}

// ---------------------------------------------------------------------------
// Router: logits = cv @ router_w.T + b, softmax, top-3, renormalize. B=4 rows.
// ---------------------------------------------------------------------------
__global__ void router_topk_kernel(const float* __restrict__ cv,
                                   const float* __restrict__ rw,
                                   const float* __restrict__ rb,
                                   int* __restrict__ topi, float* __restrict__ topw) {
  int b = threadIdx.x;
  if (b >= kB) return;
  float lg[kNEXP];
  float mx = -1e30f;
  for (int e = 0; e < kNEXP; ++e) {
    float s = rb[e];
    for (int j = 0; j < kNEXP; ++j) s += cv[b * kNEXP + j] * rw[e * kNEXP + j];
    lg[e] = s;
    mx = fmaxf(mx, s);
  }
  float sum = 0.f;
  for (int e = 0; e < kNEXP; ++e) { lg[e] = __expf(lg[e] - mx); sum += lg[e]; }
  for (int e = 0; e < kNEXP; ++e) lg[e] /= sum;
  float wsum = 0.f;
  int sel[kTOPK]; float sw[kTOPK];
  for (int i = 0; i < kTOPK; ++i) {
    int bi = 0; float bv = -1.f;
    for (int e = 0; e < kNEXP; ++e)
      if (lg[e] > bv) { bv = lg[e]; bi = e; }
    sel[i] = bi; sw[i] = bv; lg[bi] = -2.f; wsum += bv;
  }
  for (int i = 0; i < kTOPK; ++i) {
    topi[b * kTOPK + i] = sel[i];
    topw[b * kTOPK + i] = sw[i] / wsum;
  }
}

// ---------------------------------------------------------------------------
// Orchestration
// ---------------------------------------------------------------------------
extern "C" void kernel_launch(void* const* d_in, const int* in_sizes, int n_in,
                              void* d_out, int out_size, void* d_ws, size_t ws_size,
                              hipStream_t stream) {
  (void)in_sizes; (void)n_in; (void)out_size; (void)ws_size;
  const float* x        = (const float*)d_in[0];
  const float* cs       = (const float*)d_in[1];
  const float* cv       = (const float*)d_in[2];
  const float* ln_attn  = (const float*)d_in[3];
  const float* wq       = (const float*)d_in[4];
  const float* wk       = (const float*)d_in[5];
  const float* wv       = (const float*)d_in[6];
  const float* wo       = (const float*)d_in[7];
  const float* ln_pf    = (const float*)d_in[8];
  const float* pf_gate  = (const float*)d_in[9];
  const float* pf_up    = (const float*)d_in[10];
  const float* pf_down  = (const float*)d_in[11];
  const float* ln_moe   = (const float*)d_in[12];
  const float* exp_gate = (const float*)d_in[13];
  const float* exp_up   = (const float*)d_in[14];
  const float* exp_down = (const float*)d_in[15];
  const float* router_w = (const float*)d_in[16];
  const float* router_b = (const float*)d_in[17];

  float* xout    = (float*)d_out;                    // final x, also x2 staging
  float* tension = xout + (size_t)kM * kD;

  // workspace layout (~250 MB): bf16 transposed weights, bf16 activations, f32 x1
  __bf16* p = (__bf16*)d_ws;
  __bf16* wtq  = p; p += (size_t)kD * kD;
  __bf16* wtk  = p; p += (size_t)kD * kKVD;
  __bf16* wtv  = p; p += (size_t)kD * kKVD;
  __bf16* wto  = p; p += (size_t)kD * kD;
  __bf16* wtpg = p; p += (size_t)kD * kDPF;
  __bf16* wtpu = p; p += (size_t)kD * kDPF;
  __bf16* wtpd = p; p += (size_t)kDPF * kD;
  __bf16* wteg = p; p += (size_t)kNEXP * kD * kDEXP;
  __bf16* wteu = p; p += (size_t)kNEXP * kD * kDEXP;
  __bf16* wted = p; p += (size_t)kNEXP * kDEXP * kD;
  __bf16* h    = p; p += (size_t)kM * kD;
  __bf16* qb   = p; p += (size_t)kM * kD;            // q proj, then attn out
  __bf16* kb   = p; p += (size_t)kM * kKVD;
  __bf16* vb   = p; p += (size_t)kM * kKVD;
  __bf16* gb   = p; p += (size_t)kM * kDPF;          // gate / hid (in place)
  __bf16* ub   = p; p += (size_t)kM * kDPF;
  float*  x1   = (float*)p;                          // residual after attn
  int*    topi = (int*)(x1 + (size_t)kM * kD);
  float*  topw = (float*)(topi + 16);

  const dim3 blk(256);

  // ---- one-time weight transpose+convert passes ----
  transpose_cvt_kernel<<<dim3(kD/32,   kD/32,   1),     blk, 0, stream>>>(wq, wtq, kD, kD);
  transpose_cvt_kernel<<<dim3(kKVD/32, kD/32,   1),     blk, 0, stream>>>(wk, wtk, kD, kKVD);
  transpose_cvt_kernel<<<dim3(kKVD/32, kD/32,   1),     blk, 0, stream>>>(wv, wtv, kD, kKVD);
  transpose_cvt_kernel<<<dim3(kD/32,   kD/32,   1),     blk, 0, stream>>>(wo, wto, kD, kD);
  transpose_cvt_kernel<<<dim3(kDPF/32, kD/32,   1),     blk, 0, stream>>>(pf_gate, wtpg, kD, kDPF);
  transpose_cvt_kernel<<<dim3(kDPF/32, kD/32,   1),     blk, 0, stream>>>(pf_up,   wtpu, kD, kDPF);
  transpose_cvt_kernel<<<dim3(kD/32,   kDPF/32, 1),     blk, 0, stream>>>(pf_down, wtpd, kDPF, kD);
  transpose_cvt_kernel<<<dim3(kDEXP/32,kD/32,   kNEXP), blk, 0, stream>>>(exp_gate, wteg, kD, kDEXP);
  transpose_cvt_kernel<<<dim3(kDEXP/32,kD/32,   kNEXP), blk, 0, stream>>>(exp_up,   wteu, kD, kDEXP);
  transpose_cvt_kernel<<<dim3(kD/32,   kDEXP/32,kNEXP), blk, 0, stream>>>(exp_down, wted, kDEXP, kD);

  const dim3 gD (kD   / GBN, kM / GBM);
  const dim3 gKV(kKVD / GBN, kM / GBM);
  const dim3 gPF(kDPF / GBN, kM / GBM);
  const dim3 gEX(kDEXP/ GBN, kM / GBM);

  // ---- attention ----
  rmsnorm_kernel<<<kM, blk, 0, stream>>>(x, ln_attn, h, kD);
  gemm_bf16_wmma<0,false,true ><<<gD,  blk, 0, stream>>>(h, wtq, qb, kM, kD,  kD, nullptr, nullptr, nullptr, nullptr, 0, kT);
  gemm_bf16_wmma<0,false,true ><<<gKV, blk, 0, stream>>>(h, wtk, kb, kM, kKVD, kD, nullptr, nullptr, nullptr, nullptr, 0, kT);
  gemm_bf16_wmma<0,false,true ><<<gKV, blk, 0, stream>>>(h, wtv, vb, kM, kKVD, kD, nullptr, nullptr, nullptr, nullptr, 0, kT);
  flash_attn_kernel<<<dim3(kT / FQ, kH, kB), dim3(128), 0, stream>>>(qb, kb, vb, qb);
  gemm_bf16_wmma<1,false,false><<<gD,  blk, 0, stream>>>(qb, wto, x1, kM, kD, kD, x, nullptr, nullptr, nullptr, 0, kT);

  // ---- SwiGLU FFN + tension + consciousness ----
  rmsnorm_kernel<<<kM, blk, 0, stream>>>(x1, ln_pf, h, kD);
  gemm_bf16_wmma<0,false,true ><<<gPF, blk, 0, stream>>>(h, wtpg, gb, kM, kDPF, kD, nullptr, nullptr, nullptr, nullptr, 0, kT);
  gemm_bf16_wmma<0,false,true ><<<gPF, blk, 0, stream>>>(h, wtpu, ub, kM, kDPF, kD, nullptr, nullptr, nullptr, nullptr, 0, kT);
  silu_mul_kernel<<<kM, blk, 0, stream>>>(gb, ub, gb, tension, kDPF);
  gemm_bf16_wmma<2,false,false><<<gD,  blk, 0, stream>>>(gb, wtpd, xout, kM, kD, kDPF, x1, cs, nullptr, nullptr, 0, kT);

  // ---- MoE (batch-routed, top-3) ----
  rmsnorm_kernel<<<kM, blk, 0, stream>>>(xout, ln_moe, h, kD);
  router_topk_kernel<<<1, 32, 0, stream>>>(cv, router_w, router_b, topi, topw);
  for (int slot = 0; slot < kTOPK; ++slot) {
    gemm_bf16_wmma<0,true,true ><<<gEX, blk, 0, stream>>>(h, wteg, gb, kM, kDEXP, kD,
        nullptr, nullptr, topi + slot, nullptr, kTOPK, kT);
    gemm_bf16_wmma<0,true,true ><<<gEX, blk, 0, stream>>>(h, wteu, ub, kM, kDEXP, kD,
        nullptr, nullptr, topi + slot, nullptr, kTOPK, kT);
    silu_mul_kernel<<<kM, blk, 0, stream>>>(gb, ub, gb, nullptr, kDEXP);
    gemm_bf16_wmma<3,true,false><<<gD,  blk, 0, stream>>>(gb, wted, xout, kM, kD, kDEXP,
        xout, nullptr, topi + slot, topw + slot, kTOPK, kT);
  }
}
